// GCN_69389491634483
// MI455X (gfx1250) — matrix operand order repaired and checked by reading, not compiled, not run on hardware
//
#include <hip/hip_runtime.h>
#include <stdint.h>
#include <math.h>

// ---------------------------------------------------------------------------
// GCN 3-layer forward for MI455X (gfx1250), wave32 + WMMA bf16.
//   N=100000 nodes, E=1.6M edges, IN=H=128, OUT=64.
// GEMMs via v_wmma_f32_16x16x32_bf16 (A in bf16, f32 accumulate); W is
// pre-transposed to bf16 once (32KB, WGP$/L2 resident) so the GEMM needs no
// LDS staging or barriers. Message passing via one-wave-per-edge float4
// gather + global_atomic_add_f32 (51MB feature matrix is L2-resident).
// ---------------------------------------------------------------------------

typedef __bf16 bf16;
typedef __attribute__((ext_vector_type(16))) __bf16 v16bf;
typedef __attribute__((ext_vector_type(8)))  __bf16 v8bf;
typedef __attribute__((ext_vector_type(8)))  float  v8f;

// ---------------- degree / norm -------------------------------------------
__global__ __launch_bounds__(256) void init_ones(float* __restrict__ p, int n) {
  int i = blockIdx.x * 256 + threadIdx.x;
  if (i < n) p[i] = 1.0f;                       // self-loop contributes 1
}

__global__ __launch_bounds__(256) void deg_edges(const int* __restrict__ dst,
                                                 float* __restrict__ deg, int ne) {
  int i = blockIdx.x * 256 + threadIdx.x;
  if (i < ne) atomicAdd(&deg[dst[i]], 1.0f);
}

__global__ __launch_bounds__(256) void finalize_dinv(float* __restrict__ p, int n) {
  int i = blockIdx.x * 256 + threadIdx.x;
  if (i < n) p[i] = rsqrtf(p[i]);               // deg >= 1 always
}

// ---------------- f32 -> bf16 (optionally fused ReLU) ---------------------
template <bool RELU>
__global__ __launch_bounds__(256) void cvt_to_bf16(const float* __restrict__ in,
                                                   bf16* __restrict__ out, int n) {
  int i = blockIdx.x * 256 + threadIdx.x;
  if (i < n) {
    float v = in[i];
    if (RELU) v = fmaxf(v, 0.0f);
    out[i] = (bf16)v;
  }
}

// ---------------- W (f32, 128 x NC row-major) -> Wt (bf16, Wt[n*128+k]) ----
template <int NC>
__global__ __launch_bounds__(256) void pack_w_bf16(const float* __restrict__ W,
                                                   bf16* __restrict__ Wt) {
  int idx = blockIdx.x * 256 + threadIdx.x;     // over 128*NC
  if (idx < 128 * NC) {
    int k = idx / NC, n = idx - k * NC;
    Wt[n * 128 + k] = (bf16)W[idx];
  }
}

// ---------------- GEMM: T = A(bf16, Nx128) @ W(bf16^T, NCx128) -------------
// Epilogue also initializes Agg = T * dinv[row]^2 + bias[col]  (self-loop+bias)
template <int NC>
__global__ __launch_bounds__(128)
void gemm_bf16_agg(const bf16* __restrict__ A, const bf16* __restrict__ Wt,
                   const float* __restrict__ bias, const float* __restrict__ dinv,
                   float* __restrict__ T, float* __restrict__ Agg, int nrows) {
  const int tid  = threadIdx.x;
  const int wave = tid >> 5;
  const int lane = tid & 31;
  const int g    = lane >> 4;                   // lane half (ISA A/B/C layouts)
  const int l    = lane & 15;
  const int rowTile = blockIdx.x * 64 + wave * 16;
  if (rowTile >= nrows) return;                 // wave-uniform: EXEC stays full

  int rowA = rowTile + l;                       // A-matrix row M = l
  if (rowA >= nrows) rowA = nrows - 1;          // clamp loads; stores guarded

  // Preload the 4 A fragments (K tiles of 32) for this 16-row stripe.
  // 16-bit A 16x32 layout: elem e<8 -> K = kk + 8g + e ; e>=8 -> K = kk+16+8g+(e-8)
  v16bf afrag[4];
#pragma unroll
  for (int kt = 0; kt < 4; ++kt) {
    const int kk = kt * 32;
    v8bf lo = *(const v8bf*)(A + (size_t)rowA * 128 + kk + g * 8);
    v8bf hi = *(const v8bf*)(A + (size_t)rowA * 128 + kk + 16 + g * 8);
#pragma unroll
    for (int i = 0; i < 8; ++i) { afrag[kt][i] = lo[i]; afrag[kt][i + 8] = hi[i]; }
  }

  // Hoisted per-row self-loop norms dinv^2 (C/D: VGPR r -> row 8g + r).
  const bool full = (rowTile + 16) <= nrows;
  float di2[8];
#pragma unroll
  for (int r = 0; r < 8; ++r) {
    int row = rowTile + g * 8 + r;
    if (row >= nrows) row = nrows - 1;
    const float di = dinv[row];
    di2[r] = di * di;
  }

#pragma unroll
  for (int nt = 0; nt < NC / 16; ++nt) {
    const int nb  = nt * 16;
    const int col = nb + l;                     // B/C column = lane&15
    v8f acc = {};
#pragma unroll
    for (int kt = 0; kt < 4; ++kt) {
      const int kk = kt * 32;
      // 16-bit B 32x16 layout: elem e<8 -> K = kk + 8g + e ; e>=8 -> K = kk+16+8g+(e-8), col l.
      v8bf blo = *(const v8bf*)(Wt + col * 128 + kk + g * 8);
      v8bf bhi = *(const v8bf*)(Wt + col * 128 + kk + 16 + g * 8);
      v16bf bfrag = __builtin_shufflevector(blo, bhi, 0, 1, 2, 3, 4, 5, 6, 7, 8, 9, 10, 11, 12, 13, 14, 15);
      acc = __builtin_amdgcn_wmma_f32_16x16x32_bf16(
          /*neg_a=*/false, afrag[kt], /*neg_b=*/false, bfrag,
          /*c_mod=*/(short)0, acc, /*reuse_a=*/false, /*reuse_b=*/false);
    }
    const float bcol = bias[col];
    if (full) {                                 // fast path: no exec-mask churn
#pragma unroll
      for (int r = 0; r < 8; ++r) {
        const int row = rowTile + g * 8 + r;
        const size_t o = (size_t)row * NC + col;
        const float v = acc[r];
        T[o]   = v;
        Agg[o] = v * di2[r] + bcol;
      }
    } else {                                    // tail block only
#pragma unroll
      for (int r = 0; r < 8; ++r) {
        const int row = rowTile + g * 8 + r;
        if (row < nrows) {
          const size_t o = (size_t)row * NC + col;
          const float v = acc[r];
          T[o]   = v;
          Agg[o] = v * di2[r] + bcol;
        }
      }
    }
  }
}

// ---------------- edge aggregation: Agg[dst] += T[src] * dinv_s * dinv_d ---
template <int F>
__global__ __launch_bounds__(256)
void edge_aggregate(const int* __restrict__ src, const int* __restrict__ dst,
                    const float* __restrict__ dinv, const float* __restrict__ T,
                    float* __restrict__ Agg, int ne) {
  const int e    = (blockIdx.x * 256 + threadIdx.x) >> 5;   // one wave per edge
  const int lane = threadIdx.x & 31;
  if (e >= ne) return;
  const int s = src[e];
  const int d = dst[e];
  const float nrm = dinv[s] * dinv[d];
  constexpr int PER = F / 32;                   // 4 for F=128, 2 for F=64
  const float* tp = T   + (size_t)s * F + lane * PER;
  float*       ap = Agg + (size_t)d * F + lane * PER;
  if constexpr (PER == 4) {
    float4 v = *(const float4*)tp;
    atomicAdd(ap + 0, v.x * nrm);
    atomicAdd(ap + 1, v.y * nrm);
    atomicAdd(ap + 2, v.z * nrm);
    atomicAdd(ap + 3, v.w * nrm);
  } else {
    float2 v = *(const float2*)tp;
    atomicAdd(ap + 0, v.x * nrm);
    atomicAdd(ap + 1, v.y * nrm);
  }
}

// ---------------- log_softmax over 64 outputs (one wave per node) ----------
__global__ __launch_bounds__(256)
void logsoftmax64(const float* __restrict__ in, float* __restrict__ out, int n) {
  const int node = (blockIdx.x * 256 + threadIdx.x) >> 5;
  const int lane = threadIdx.x & 31;
  if (node >= n) return;
  const float* row = in + (size_t)node * 64;
  float a = row[lane];
  float b = row[lane + 32];
  float m = fmaxf(a, b);
#pragma unroll
  for (int off = 16; off; off >>= 1) m = fmaxf(m, __shfl_xor(m, off, 32));
  float s = expf(a - m) + expf(b - m);
#pragma unroll
  for (int off = 16; off; off >>= 1) s += __shfl_xor(s, off, 32);
  const float lse = m + logf(s);
  float* o = out + (size_t)node * 64;
  o[lane]      = a - lse;
  o[lane + 32] = b - lse;
}

// ---------------------------------------------------------------------------
extern "C" void kernel_launch(void* const* d_in, const int* in_sizes, int n_in,
                              void* d_out, int out_size, void* d_ws, size_t ws_size,
                              hipStream_t stream) {
  const float* x  = (const float*)d_in[0];
  const int*   ei = (const int*)d_in[1];      // [2, E] int32
  const float* W1 = (const float*)d_in[2];
  const float* b1 = (const float*)d_in[3];
  const float* W2 = (const float*)d_in[4];
  const float* b2 = (const float*)d_in[5];
  const float* W3 = (const float*)d_in[6];
  const float* b3 = (const float*)d_in[7];

  const int N = in_sizes[0] / 128;
  const int E = in_sizes[1] / 2;
  const int* srcp = ei;
  const int* dstp = ei + E;

  // Workspace carve-up (256B aligned):
  //   dinv: N f32 | Wt: 128*128 bf16 | A16: N*128 bf16 | T: N*128 f32 | Agg: N*128 f32
  auto align256 = [](size_t v) { return (v + 255) & ~(size_t)255; };
  uintptr_t base = (uintptr_t)d_ws;
  size_t off = 0;
  float* dinv = (float*)(base + off); off = align256(off + (size_t)N * 4);
  bf16*  Wt   = (bf16*)(base + off);  off = align256(off + (size_t)128 * 128 * 2);
  bf16*  A16  = (bf16*)(base + off);  off = align256(off + (size_t)N * 128 * 2);
  float* T    = (float*)(base + off); off = align256(off + (size_t)N * 128 * 4);
  float* Agg  = (float*)(base + off); off = align256(off + (size_t)N * 128 * 4);
  float* out  = (float*)d_out;

  auto cdiv = [](int a, int b) { return (a + b - 1) / b; };
  const int gN    = cdiv(N, 256);
  const int gE    = cdiv(E, 256);
  const int gNF   = cdiv(N * 128, 256);
  const int gGemm = cdiv(N, 64);
  const int gEw   = cdiv(E, 8);               // 8 waves / 256-thread block
  const int gSm   = cdiv(N, 8);
  const int gW128 = cdiv(128 * 128, 256);
  const int gW64  = cdiv(128 * 64, 256);

  // --- degree normalization (shared by all 3 layers) ---
  init_ones<<<gN, 256, 0, stream>>>(dinv, N);
  deg_edges<<<gE, 256, 0, stream>>>(dstp, dinv, E);
  finalize_dinv<<<gN, 256, 0, stream>>>(dinv, N);

  // --- layer 1: h1 = relu(Agg(x@W1) + b1) ---
  cvt_to_bf16<false><<<gNF, 256, 0, stream>>>(x, A16, N * 128);
  pack_w_bf16<128><<<gW128, 256, 0, stream>>>(W1, Wt);
  gemm_bf16_agg<128><<<gGemm, 128, 0, stream>>>(A16, Wt, b1, dinv, T, Agg, N);
  edge_aggregate<128><<<gEw, 256, 0, stream>>>(srcp, dstp, dinv, T, Agg, E);

  // --- layer 2 ---
  cvt_to_bf16<true><<<gNF, 256, 0, stream>>>(Agg, A16, N * 128);
  pack_w_bf16<128><<<gW128, 256, 0, stream>>>(W2, Wt);
  gemm_bf16_agg<128><<<gGemm, 128, 0, stream>>>(A16, Wt, b2, dinv, T, Agg, N);
  edge_aggregate<128><<<gEw, 256, 0, stream>>>(srcp, dstp, dinv, T, Agg, E);

  // --- layer 3 (OUT=64, no relu) + log_softmax ---
  cvt_to_bf16<true><<<gNF, 256, 0, stream>>>(Agg, A16, N * 128);
  pack_w_bf16<64><<<gW64, 256, 0, stream>>>(W3, Wt);
  gemm_bf16_agg<64><<<gGemm, 128, 0, stream>>>(A16, Wt, b3, dinv, T, Agg, N);
  edge_aggregate<64><<<gEw, 256, 0, stream>>>(srcp, dstp, dinv, T, Agg, E);
  logsoftmax64<<<gSm, 256, 0, stream>>>(Agg, out, N);
}
